// PrivacyDetectionUnit_79018808312355
// MI455X (gfx1250) — matrix operand order repaired and checked by reading, not compiled, not run on hardware
//
#include <hip/hip_runtime.h>
#include <hip/hip_bf16.h>

typedef _Float16 v8h  __attribute__((ext_vector_type(8)));
typedef _Float16 v16h __attribute__((ext_vector_type(16)));
typedef float    v8f  __attribute__((ext_vector_type(8)));

static constexpr int N_NODES = 10000;
static constexpr int DIM     = 512;
static constexpr int NB      = 64;
static constexpr int NE      = 160000;
static constexpr int ET      = NE + N_NODES;   // edges + self loops

#define DEVFN static __device__ __forceinline__

// ---- orderable-uint encoding for float atomic max ----
DEVFN unsigned f2key(float f){ unsigned u = __float_as_uint(f); return (u & 0x80000000u) ? ~u : (u | 0x80000000u); }
DEVFN float key2f(unsigned k){ unsigned u = (k & 0x80000000u) ? (k ^ 0x80000000u) : ~k; return __uint_as_float(u); }

DEVFN void edge_sd(const int* __restrict__ ei, int e, int& s, int& d){
  if (e < NE){ s = ei[e]; d = ei[NE + e]; } else { s = d = e - NE; }  // appended self-loops
}

// ------------------------------------------------------------------
// utility kernels
// ------------------------------------------------------------------
__global__ void k_f32_to_f16(const float* __restrict__ in, _Float16* __restrict__ out, int n){
  int i = blockIdx.x * blockDim.x + threadIdx.x;
  if (i < n) out[i] = (_Float16)in[i];
}

// W[rows,cols] (f32) -> Wt[cols,rows] (f16)
__global__ void k_transpose_f16(const float* __restrict__ W, _Float16* __restrict__ Wt, int rows, int cols){
  int i = blockIdx.x * blockDim.x + threadIdx.x;
  if (i >= rows * cols) return;
  int r = i / cols, c = i % cols;
  Wt[(size_t)c * rows + r] = (_Float16)W[i];
}

__global__ void k_fill_u32(unsigned* p, unsigned v, int n){
  int i = blockIdx.x * blockDim.x + threadIdx.x;
  if (i < n) p[i] = v;
}
__global__ void k_fill_f32(float* p, float v, int n){
  int i = blockIdx.x * blockDim.x + threadIdx.x;
  if (i < n) p[i] = v;
}
// out[r, c] = bias[c]  for r in [0, rows)
__global__ void k_bias_bcast(const float* __restrict__ b, float* __restrict__ out, int rows, int cols){
  int i = blockIdx.x * blockDim.x + threadIdx.x;
  if (i >= rows * cols) return;
  out[i] = b[i % cols];
}

// ------------------------------------------------------------------
// WMMA GEMM: C[rows,512] (f32) = A[rows,512] (f16) * Bt^T, Bt[512,512] f16 row-major (= W^T).
// Block (256 thr / 8 waves) owns one 64-col strip of Bt: staged to LDS once via
// async-to-LDS loads (ASYNCcnt), then all waves hit LDS for B fragments.
// Wave computes a 16x64 strip (4 accumulators); A fragments software-pipelined.
// ------------------------------------------------------------------
__global__ __launch_bounds__(256) void k_gemm_wmma(
    const _Float16* __restrict__ A, const _Float16* __restrict__ Bt,
    float* __restrict__ C, int rows)
{
  const int K = DIM;
  __shared__ _Float16 sB[64 * DIM];          // 64 cols x 512 K x 2B = 64 KB

  int cs   = blockIdx.x & 7;                 // column strip (64 cols)
  int rb   = blockIdx.x >> 3;                // row block (8 row tiles)
  int wv   = threadIdx.x >> 5;
  int lane = threadIdx.x & 31;
  int rt   = rb * 8 + wv;                    // this wave's 16-row tile

  // ---- stage the contiguous 64KB strip Bt[cs*64 .. cs*64+63][*] into LDS ----
  {
    const _Float16* src = Bt + (size_t)cs * 64 * K;
    int base = threadIdx.x * 128;            // 128 halfs = 256 B per thread
    unsigned lds0 = (unsigned)(size_t)&sB[base];
    unsigned long long g0 = (unsigned long long)(const void*)(src + base);
    #pragma unroll
    for (int i = 0; i < 16; ++i){
      unsigned laddr = lds0 + (unsigned)(i * 16);
      unsigned long long gaddr = g0 + (unsigned long long)(i * 16);
      asm volatile("global_load_async_to_lds_b128 %0, %1, off"
                   :: "v"(laddr), "v"(gaddr) : "memory");
    }
    asm volatile("s_wait_asynccnt 0x0" ::: "memory");
  }
  __syncthreads();
  if (rt * 16 >= rows) return;

  int l15 = lane & 15, hi = lane >> 4;
  const _Float16* arow = A + (size_t)(rt * 16 + l15) * K;
  const _Float16* sb0 = &sB[(size_t)( 0 + l15) * K];
  const _Float16* sb1 = &sB[(size_t)(16 + l15) * K];
  const _Float16* sb2 = &sB[(size_t)(32 + l15) * K];
  const _Float16* sb3 = &sB[(size_t)(48 + l15) * K];

  v8f acc0 = {}, acc1 = {}, acc2 = {}, acc3 = {};
  union U { v16h v; v8h h[2]; };

  // prime the A-fragment pipeline (16-bit A 16x32 layout: lane=M, K = [8hi..+7] u [16+8hi..+7])
  U a;
  {
    int oa = 8 * hi;
    a.h[0] = *(const v8h*)(arow + oa);
    a.h[1] = *(const v8h*)(arow + oa + 16);
  }
  for (int k0 = 0; k0 < K; k0 += 32){
    U acur = a;
    if (k0 + 32 < K){                        // prefetch next A fragment (global)
      int oa = k0 + 32 + 8 * hi;
      a.h[0] = *(const v8h*)(arow + oa);
      a.h[1] = *(const v8h*)(arow + oa + 16);
    }
    // B fragments from LDS: lane=N, K = [8*hi .. +7] u [16+8*hi .. +7]
    int ob = k0 + 8 * hi;
    U f0, f1, f2, f3;
    f0.h[0] = *(const v8h*)(sb0 + ob); f0.h[1] = *(const v8h*)(sb0 + ob + 16);
    f1.h[0] = *(const v8h*)(sb1 + ob); f1.h[1] = *(const v8h*)(sb1 + ob + 16);
    f2.h[0] = *(const v8h*)(sb2 + ob); f2.h[1] = *(const v8h*)(sb2 + ob + 16);
    f3.h[0] = *(const v8h*)(sb3 + ob); f3.h[1] = *(const v8h*)(sb3 + ob + 16);

    acc0 = __builtin_amdgcn_wmma_f32_16x16x32_f16(false, acur.v, false, f0.v, (short)0, acc0, false, false);
    acc1 = __builtin_amdgcn_wmma_f32_16x16x32_f16(false, acur.v, false, f1.v, (short)0, acc1, false, false);
    acc2 = __builtin_amdgcn_wmma_f32_16x16x32_f16(false, acur.v, false, f2.v, (short)0, acc2, false, false);
    acc3 = __builtin_amdgcn_wmma_f32_16x16x32_f16(false, acur.v, false, f3.v, (short)0, acc3, false, false);
  }
  // C/D layout: lane l15 = column, VGPR r -> row 8*hi + r
  int col0 = cs * 64;
  for (int r = 0; r < 8; ++r){
    float* crow = C + (size_t)(rt * 16 + hi * 8 + r) * DIM + col0 + l15;
    crow[ 0] = acc0[r];
    crow[16] = acc1[r];
    crow[32] = acc2[r];
    crow[48] = acc3[r];
  }
}

// ------------------------------------------------------------------
// GAT pieces
// ------------------------------------------------------------------
// per-(node, head) attention scores: s = sum_c xp[n, h*oc + c] * a[h*oc + c]
__global__ void k_scores(const float* __restrict__ xp, const float* __restrict__ a_src,
                         const float* __restrict__ a_dst, float* __restrict__ ss,
                         float* __restrict__ sd, int H, int oc){
  int t = blockIdx.x * blockDim.x + threadIdx.x;
  if (t >= N_NODES * H) return;
  int n = t / H, h = t % H;
  const float* row = xp + (size_t)n * DIM + h * oc;
  const float* as = a_src + h * oc;
  const float* ad = a_dst + h * oc;
  float s0 = 0.f, s1 = 0.f;
  for (int c = 0; c < oc; ++c){ float v = row[c]; s0 += v * as[c]; s1 += v * ad[c]; }
  ss[t] = s0; sd[t] = s1;
}

__global__ void k_edge_max(const int* __restrict__ ei, const float* __restrict__ ss,
                           const float* __restrict__ sd, unsigned* __restrict__ mk, int H){
  int t = blockIdx.x * blockDim.x + threadIdx.x;
  if (t >= ET * H) return;
  int e = t / H, h = t % H;
  int s, d; edge_sd(ei, e, s, d);
  float sc = ss[s * H + h] + sd[d * H + h];
  sc = sc > 0.f ? sc : 0.2f * sc;              // leaky_relu, slope 0.2
  atomicMax(&mk[d * H + h], f2key(sc));
}

__global__ void k_edge_expsum(const int* __restrict__ ei, const float* __restrict__ ss,
                              const float* __restrict__ sd, const unsigned* __restrict__ mk,
                              float* __restrict__ ee, float* __restrict__ dn, int H){
  int t = blockIdx.x * blockDim.x + threadIdx.x;
  if (t >= ET * H) return;
  int e = t / H, h = t % H;
  int s, d; edge_sd(ei, e, s, d);
  float sc = ss[s * H + h] + sd[d * H + h];
  sc = sc > 0.f ? sc : 0.2f * sc;
  float v = expf(sc - key2f(mk[d * H + h]));
  ee[e * H + h] = v;
  atomicAdd(&dn[d * H + h], v);
}

// out[dst] += alpha * xp[src], per (edge, 4-channel chunk)
__global__ void k_aggregate(const int* __restrict__ ei, const float* __restrict__ xp,
                            const float* __restrict__ ee, const float* __restrict__ dn,
                            float* __restrict__ out, int H, int oc){
  int t = blockIdx.x * blockDim.x + threadIdx.x;
  const int CH = DIM / 4;
  if (t >= ET * CH) return;
  int e = t / CH;
  int j = (t % CH) * 4;
  int s, d; edge_sd(ei, e, s, d);
  int h = j / oc;
  float alpha = ee[e * H + h] / dn[d * H + h];
  float4 xv = *(const float4*)(xp + (size_t)s * DIM + j);
  float* op = out + (size_t)d * DIM + j;
  atomicAdd(op + 0, xv.x * alpha);
  atomicAdd(op + 1, xv.y * alpha);
  atomicAdd(op + 2, xv.z * alpha);
  atomicAdd(op + 3, xv.w * alpha);
}

// ELU then convert to f16 (feeds layer-2 GEMM)
__global__ void k_elu_f16(const float* __restrict__ in, _Float16* __restrict__ out, int n){
  int i = blockIdx.x * blockDim.x + threadIdx.x;
  if (i >= n) return;
  float x = in[i];
  out[i] = (_Float16)(x > 0.f ? x : (expf(x) - 1.f));
}

// ------------------------------------------------------------------
// pooling + head
// ------------------------------------------------------------------
__global__ void k_pool(const int* __restrict__ batch, const float* __restrict__ h,
                       float* __restrict__ sums){
  int t = blockIdx.x * blockDim.x + threadIdx.x;
  const int CH = DIM / 4;
  if (t >= N_NODES * CH) return;
  int n = t / CH; int j = (t % CH) * 4;
  int b = batch[n];
  float4 v = *(const float4*)(h + (size_t)n * DIM + j);
  float* sp = sums + (size_t)b * DIM + j;
  atomicAdd(sp + 0, v.x); atomicAdd(sp + 1, v.y);
  atomicAdd(sp + 2, v.z); atomicAdd(sp + 3, v.w);
}
__global__ void k_count(const int* __restrict__ batch, float* __restrict__ cnt){
  int n = blockIdx.x * blockDim.x + threadIdx.x;
  if (n < N_NODES) atomicAdd(&cnt[batch[n]], 1.f);
}
__global__ void k_gmean(const float* __restrict__ sums, const float* __restrict__ cnt,
                        float* __restrict__ g){
  int t = blockIdx.x * blockDim.x + threadIdx.x;
  if (t >= NB * DIM) return;
  float c = cnt[t / DIM]; c = c > 1.f ? c : 1.f;
  g[t] = sums[t] / c;
}

// Y[r,c] = act( bias[c] + sum_k X[r,k] * W[k,c] )
__global__ void k_linear(const float* __restrict__ X, const float* __restrict__ W,
                         const float* __restrict__ bias, float* __restrict__ Y,
                         int rows, int kin, int kout, int relu){
  int t = blockIdx.x * blockDim.x + threadIdx.x;
  if (t >= rows * kout) return;
  int r = t / kout, c = t % kout;
  const float* x = X + (size_t)r * kin;
  float acc = bias[c];
  for (int k = 0; k < kin; ++k) acc += x[k] * W[(size_t)k * kout + c];
  if (relu && acc < 0.f) acc = 0.f;
  Y[t] = acc;
}

__global__ void k_final(const float* __restrict__ c1, const float* __restrict__ Wc2,
                        const float* __restrict__ bc2, const float* __restrict__ rule,
                        float* __restrict__ out){
  int b = blockIdx.x * blockDim.x + threadIdx.x;
  if (b >= NB) return;
  float z = bc2[0];
  for (int k = 0; k < 256; ++k) z += c1[b * 256 + k] * Wc2[k];
  float sg = 1.f / (1.f + expf(-z));
  out[b] = 0.5f * sg + 0.5f * rule[b];
}

// ------------------------------------------------------------------
// host launcher
// ------------------------------------------------------------------
static inline int cdiv(long long a, long long b){ return (int)((a + b - 1) / b); }

extern "C" void kernel_launch(void* const* d_in, const int* in_sizes, int n_in,
                              void* d_out, int out_size, void* d_ws, size_t ws_size,
                              hipStream_t stream) {
  (void)in_sizes; (void)n_in; (void)out_size; (void)ws_size;
  const float* x        = (const float*)d_in[1];
  const float* rule     = (const float*)d_in[2];
  const int*   ei       = (const int*)  d_in[3];
  const int*   batch    = (const int*)  d_in[4];
  const float* W_g1     = (const float*)d_in[7];
  const float* a_src1   = (const float*)d_in[8];
  const float* a_dst1   = (const float*)d_in[9];
  const float* b_g1     = (const float*)d_in[10];
  const float* W_g2     = (const float*)d_in[11];
  const float* a_src2   = (const float*)d_in[12];
  const float* a_dst2   = (const float*)d_in[13];
  const float* b_g2     = (const float*)d_in[14];
  const float* Wv       = (const float*)d_in[19];
  const float* bv       = (const float*)d_in[20];
  const float* Wo       = (const float*)d_in[21];
  const float* bo       = (const float*)d_in[22];
  const float* Wc1      = (const float*)d_in[23];
  const float* bc1      = (const float*)d_in[24];
  const float* Wc2      = (const float*)d_in[25];
  const float* bc2      = (const float*)d_in[26];
  float* out = (float*)d_out;          // [64] final  ++ [64*512] attn_out

  // workspace carve-up (256B aligned)
  char* base = (char*)d_ws;
  size_t off = 0;
  auto carve = [&](size_t bytes) -> char* {
    char* p = base + off;
    off = (off + bytes + 255) & ~(size_t)255;
    return p;
  };
  _Float16* xh   = (_Float16*)carve((size_t)N_NODES * DIM * 2);   // f16 node features
  _Float16* Wt   = (_Float16*)carve((size_t)DIM * DIM * 2);       // f16 transposed weights
  float*    xp   = (float*)   carve((size_t)N_NODES * DIM * 4);   // post-GEMM features
  float*    hb   = (float*)   carve((size_t)N_NODES * DIM * 4);   // aggregated output
  float*    ss   = (float*)   carve((size_t)N_NODES * 4 * 4);
  float*    sd   = (float*)   carve((size_t)N_NODES * 4 * 4);
  unsigned* mk   = (unsigned*)carve((size_t)N_NODES * 4 * 4);
  float*    dn   = (float*)   carve((size_t)N_NODES * 4 * 4);
  float*    ee   = (float*)   carve((size_t)ET * 4 * 4);
  float*    sums = (float*)   carve((size_t)NB * DIM * 4);
  float*    cnt  = (float*)   carve((size_t)NB * 4);
  float*    gb   = (float*)   carve((size_t)NB * DIM * 4);
  float*    vb   = (float*)   carve((size_t)NB * DIM * 4);
  float*    c1   = (float*)   carve((size_t)NB * 256 * 4);

  const int T = 256;
  const int nND  = N_NODES * DIM;            // 5.12M
  const int nWW  = DIM * DIM;                // 262144
  const int nAGG = ET * (DIM / 4);           // 21.76M
  const int row_tiles = N_NODES / 16;        // 625
  const int gemm_blocks = cdiv(row_tiles, 8) * 8;   // 79 row blocks * 8 col strips = 632

  // ---------------- GAT layer 1 (H=4, oc=128) ----------------
  k_f32_to_f16   <<<cdiv(nND, T), T, 0, stream>>>(x, xh, nND);
  k_transpose_f16<<<cdiv(nWW, T), T, 0, stream>>>(W_g1, Wt, DIM, DIM);
  k_gemm_wmma    <<<gemm_blocks, T, 0, stream>>>(xh, Wt, xp, N_NODES);
  k_scores       <<<cdiv(N_NODES * 4, T), T, 0, stream>>>(xp, a_src1, a_dst1, ss, sd, 4, 128);
  k_fill_u32     <<<cdiv(N_NODES * 4, T), T, 0, stream>>>(mk, 0u, N_NODES * 4);
  k_fill_f32     <<<cdiv(N_NODES * 4, T), T, 0, stream>>>(dn, 0.f, N_NODES * 4);
  k_edge_max     <<<cdiv((long long)ET * 4, T), T, 0, stream>>>(ei, ss, sd, mk, 4);
  k_edge_expsum  <<<cdiv((long long)ET * 4, T), T, 0, stream>>>(ei, ss, sd, mk, ee, dn, 4);
  k_bias_bcast   <<<cdiv(nND, T), T, 0, stream>>>(b_g1, hb, N_NODES, DIM);
  k_aggregate    <<<cdiv(nAGG, T), T, 0, stream>>>(ei, xp, ee, dn, hb, 4, 128);

  // ---------------- ELU -> f16, GAT layer 2 (H=1, oc=512) ----------------
  k_elu_f16      <<<cdiv(nND, T), T, 0, stream>>>(hb, xh, nND);
  k_transpose_f16<<<cdiv(nWW, T), T, 0, stream>>>(W_g2, Wt, DIM, DIM);
  k_gemm_wmma    <<<gemm_blocks, T, 0, stream>>>(xh, Wt, xp, N_NODES);
  k_scores       <<<cdiv(N_NODES, T), T, 0, stream>>>(xp, a_src2, a_dst2, ss, sd, 1, 512);
  k_fill_u32     <<<cdiv(N_NODES, T), T, 0, stream>>>(mk, 0u, N_NODES);
  k_fill_f32     <<<cdiv(N_NODES, T), T, 0, stream>>>(dn, 0.f, N_NODES);
  k_edge_max     <<<cdiv(ET, T), T, 0, stream>>>(ei, ss, sd, mk, 1);
  k_edge_expsum  <<<cdiv(ET, T), T, 0, stream>>>(ei, ss, sd, mk, ee, dn, 1);
  k_bias_bcast   <<<cdiv(nND, T), T, 0, stream>>>(b_g2, hb, N_NODES, DIM);
  k_aggregate    <<<cdiv(nAGG, T), T, 0, stream>>>(ei, xp, ee, dn, hb, 1, 512);

  // ---------------- global mean pool ----------------
  k_fill_f32     <<<cdiv(NB * DIM, T), T, 0, stream>>>(sums, 0.f, NB * DIM);
  k_fill_f32     <<<1, NB, 0, stream>>>(cnt, 0.f, NB);
  k_pool         <<<cdiv(N_NODES * (DIM / 4), T), T, 0, stream>>>(batch, hb, sums);
  k_count        <<<cdiv(N_NODES, T), T, 0, stream>>>(batch, cnt);
  k_gmean        <<<cdiv(NB * DIM, T), T, 0, stream>>>(sums, cnt, gb);

  // ---------------- attention (seq len 1 -> just v/out proj) + classifier ----------------
  k_linear       <<<cdiv(NB * DIM, T), T, 0, stream>>>(gb, Wv, bv, vb, NB, DIM, DIM, 0);
  k_linear       <<<cdiv(NB * DIM, T), T, 0, stream>>>(vb, Wo, bo, out + NB, NB, DIM, DIM, 0); // attn_out
  k_linear       <<<cdiv(NB * 256, T), T, 0, stream>>>(out + NB, Wc1, bc1, c1, NB, DIM, 256, 1);
  k_final        <<<1, NB, 0, stream>>>(c1, Wc2, bc2, rule, out);
}
